// ShiftBlock_51719996178709
// MI455X (gfx1250) — 8-bit matrix operand order repaired and checked by reading, compiled, not run on hardware
//
#include <hip/hip_runtime.h>


typedef int   v8i __attribute__((ext_vector_type(8)));
typedef int   v4i __attribute__((ext_vector_type(4)));
typedef int   v2i __attribute__((ext_vector_type(2)));
typedef float v8f __attribute__((ext_vector_type(8)));

#define BATCH 32
#define CH    256
#define HH    56
#define WW    56
#define HWS   3136
#define NPIX  100352

#define WS_SCALES   0
#define WS_GSUM     3072
#define WS_GSQ      4096
#define WS_BNSCALE  5120
#define WS_BNSHIFT  6144
#define WS_PREPB    8192
#define WS_A        204800
#define WS_S        25894912

__global__ void prep_kernel(const float* __restrict__ w1,
                            const float* __restrict__ w2,
                            const float* __restrict__ w3,
                            float* __restrict__ scales,
                            float* __restrict__ gsum,
                            float* __restrict__ gsq,
                            int* __restrict__ prepB)
{
    const int conv = blockIdx.x;
    const int t    = threadIdx.x;
    const float* w = (conv == 0) ? w1 : ((conv == 1) ? w2 : w3);

    if (conv == 0) { gsum[t] = 0.0f; gsq[t] = 0.0f; }

    float s = 0.0f;
    for (int c = 0; c < 256; ++c) s += fabsf(w[t * 256 + c]);
    scales[conv * 256 + t] = s * (1.0f / 256.0f);

    int* pb = prepB + conv * 16384;
    for (int d = 0; d < 64; ++d) {
        int gid  = t * 64 + d;
        int dw   = gid & 7;
        int lane = (gid >> 3) & 31;
        int nt   = (gid >> 8) & 15;
        int kk   = (gid >> 12) & 3;
        int col  = nt * 16 + (lane & 15);
        int kb   = kk * 64 + (dw >> 1) * 16 + ((lane & 16) ? 8 : 0) + (dw & 1) * 4;
        unsigned pk = 0u;
        for (int j = 0; j < 4; ++j) {
            float wv = w[col * 256 + kb + j];
            int sg = (wv > 0.0f) ? 1 : ((wv < 0.0f) ? -1 : 0);
            pk |= ((unsigned)(sg & 0xFF)) << (8 * j);
        }
        pb[gid] = (int)pk;
    }
}

__global__ void act_kernel(const float* __restrict__ x,
                           const float* __restrict__ bias0,
                           char* __restrict__ a)
{
    __shared__ char lds8[14336];
    const int bh = blockIdx.x;
    const int b  = bh / HH;
    const int h  = bh - b * HH;
    const int t  = threadIdx.x;

    for (int i = t; i < 14336; i += 256) {
        int c = i & 255;
        int w = i >> 8;
        float v = x[(size_t)(b * 256 + c) * HWS + h * WW + w] + bias0[c];
        lds8[i] = (v > 0.0f) ? (char)1 : ((v < 0.0f) ? (char)-1 : (char)0);
    }
    __syncthreads();

    int4* dst       = (int4*)(a + (size_t)bh * 14336);
    const int4* src = (const int4*)lds8;
    for (int i = t; i < 896; i += 256) dst[i] = src[i];
}

__global__ void __launch_bounds__(256)
__attribute__((amdgpu_waves_per_eu(2)))
gemm_kernel(const char*  __restrict__ a,
            const int*   __restrict__ prepB,
            const float* __restrict__ scales,
            float* __restrict__ s_out,
            float* __restrict__ gsum,
            float* __restrict__ gsq)
{
    __shared__ int   ldsB[12288];
    __shared__ float lds_sum[64];
    __shared__ float lds_sq[64];

    const int t    = threadIdx.x;
    const int wv   = t >> 5;
    const int lane = t & 31;

    const int pgrp = blockIdx.x >> 2;
    const int ntq  = blockIdx.x & 3;

    {
        const int4* gB = (const int4*)prepB;
        int4* lB = (int4*)ldsB;
        for (int i4 = t; i4 < 3072; i4 += 256) {
            int chunk = i4 >> 8;
            int within = i4 & 255;
            int conv = chunk >> 2;
            int kk   = chunk & 3;
            lB[i4] = gB[conv * 4096 + (kk * 16 + ntq * 4) * 64 + within];
        }
    }
    if (t < 64) { lds_sum[t] = 0.0f; lds_sq[t] = 0.0f; }
    __syncthreads();

    const int pt   = pgrp * 8 + wv;
    const int row  = lane & 15;
    const int half = lane >> 4;
    const int koff = half << 3;

    const int p   = pt * 16 + row;
    const int b   = p / HWS;
    const int rem = p - b * HWS;
    const int h   = rem / WW;
    const int w   = rem - h * WW;

    int psrc[3][4];
#pragma unroll
    for (int conv = 0; conv < 3; ++conv) {
        int k  = (conv == 0) ? 0 : ((conv == 1) ? 1 : 3);
        int hm = h - k; if (hm < 0)   hm += HH;
        int hp = h + k; if (hp >= HH) hp -= HH;
        int wm = w - k; if (wm < 0)   wm += WW;
        int wp = w + k; if (wp >= WW) wp -= WW;
        psrc[conv][0] = b * HWS + hm * WW + w;
        psrc[conv][1] = b * HWS + hp * WW + w;
        psrc[conv][2] = b * HWS + h * WW + wm;
        psrc[conv][3] = b * HWS + h * WW + wp;
    }

    v8i Am[3][4];
#pragma unroll
    for (int conv = 0; conv < 3; ++conv) {
#pragma unroll
        for (int kk = 0; kk < 4; ++kk) {
            const char* ap = a + (size_t)psrc[conv][kk] * 256 + kk * 64 + koff;
            v2i g0 = *(const v2i*)(ap);
            v2i g1 = *(const v2i*)(ap + 16);
            v2i g2 = *(const v2i*)(ap + 32);
            v2i g3 = *(const v2i*)(ap + 48);
            v8i Av;
            Av[0] = g0[0]; Av[1] = g0[1]; Av[2] = g1[0]; Av[3] = g1[1];
            Av[4] = g2[0]; Av[5] = g2[1]; Av[6] = g3[0]; Av[7] = g3[1];
            Am[conv][kk] = Av;
        }
    }

    const int p0   = pt * 16;
    const int b0   = p0 / HWS;
    const int inb0 = p0 - b0 * HWS;

#pragma unroll
    for (int ntl = 0; ntl < 4; ++ntl) {
        const int cbase = (ntq * 4 + ntl) * 16;
        v8f facc = {};
#pragma unroll
        for (int conv = 0; conv < 3; ++conv) {
            v8i acc = {};
#pragma unroll
            for (int kk = 0; kk < 4; ++kk) {
                const int* bp = ldsB + ((((conv * 4 + kk) * 4 + ntl) * 32 + lane) * 8);
                v4i q0 = *(const v4i*)(bp);
                v4i q1 = *(const v4i*)(bp + 4);
                v8i Bm;
                Bm[0] = q0[0]; Bm[1] = q0[1]; Bm[2] = q0[2]; Bm[3] = q0[3];
                Bm[4] = q1[0]; Bm[5] = q1[1]; Bm[6] = q1[2]; Bm[7] = q1[3];

                acc = __builtin_amdgcn_wmma_i32_16x16x64_iu8(
                        true, Am[conv][kk], true, Bm, acc, false, false);
            }
            float sc = scales[conv * 256 + cbase + (lane & 15)];
#pragma unroll
            for (int r = 0; r < 8; ++r) facc[r] += sc * (float)acc[r];
        }

        float ls = 0.0f, lq = 0.0f;
#pragma unroll
        for (int r = 0; r < 8; ++r) { float v = facc[r]; ls += v; lq += v * v; }
        atomicAdd(&lds_sum[ntl * 16 + (lane & 15)], ls);
        atomicAdd(&lds_sq [ntl * 16 + (lane & 15)], lq);

        float* sp = s_out + (size_t)(b0 * 256 + cbase + (lane & 15)) * HWS
                          + inb0 + half * 8;
        *(float4*)(sp)     = make_float4(facc[0], facc[1], facc[2], facc[3]);
        *(float4*)(sp + 4) = make_float4(facc[4], facc[5], facc[6], facc[7]);
    }

    __syncthreads();
    if (t < 64) {
        atomicAdd(&gsum[ntq * 64 + t], lds_sum[t]);
        atomicAdd(&gsq [ntq * 64 + t], lds_sq[t]);
    }
}

__global__ void bnfin_kernel(const float* __restrict__ gsum,
                             const float* __restrict__ gsq,
                             const float* __restrict__ gamma,
                             const float* __restrict__ beta,
                             float* __restrict__ bnscale,
                             float* __restrict__ bnshift)
{
    const int t = threadIdx.x;
    const float n    = (float)NPIX;
    const float mean = gsum[t] / n;
    const float var  = gsq[t] / n - mean * mean;
    const float inv  = rsqrtf(var + 1e-5f);
    const float sc   = gamma[t] * inv;
    bnscale[t] = sc;
    bnshift[t] = beta[t] - mean * sc;
}

__global__ void out_kernel(const float* __restrict__ s,
                           const float* __restrict__ x,
                           const float* __restrict__ bnscale,
                           const float* __restrict__ bnshift,
                           const float* __restrict__ bias1,
                           const float* __restrict__ alpha,
                           const float* __restrict__ bias2,
                           float* __restrict__ out)
{
    const int i4 = blockIdx.x * 256 + threadIdx.x;
    if (i4 >= (NPIX * 256 / 4)) return;
    const int c = (i4 / (HWS / 4)) & 255;

    const float4 sv = ((const float4*)s)[i4];
    const float4 xv = ((const float4*)x)[i4];
    const float bs = bnscale[c], bh = bnshift[c];
    const float b1 = bias1[c],   al = alpha[c], b2 = bias2[c];

    float4 o;
    {
        float hv = sv.x * bs + bh + xv.x + b1;
        o.x = ((hv >= 0.0f) ? hv : al * hv) + b2;
    }
    {
        float hv = sv.y * bs + bh + xv.y + b1;
        o.y = ((hv >= 0.0f) ? hv : al * hv) + b2;
    }
    {
        float hv = sv.z * bs + bh + xv.z + b1;
        o.z = ((hv >= 0.0f) ? hv : al * hv) + b2;
    }
    {
        float hv = sv.w * bs + bh + xv.w + b1;
        o.w = ((hv >= 0.0f) ? hv : al * hv) + b2;
    }
    ((float4*)out)[i4] = o;
}

extern "C" void kernel_launch(void* const* d_in, const int* in_sizes, int n_in,
                              void* d_out, int out_size, void* d_ws, size_t ws_size,
                              hipStream_t stream)
{
    const float* x     = (const float*)d_in[0];
    const float* bias0 = (const float*)d_in[1];
    const float* w1    = (const float*)d_in[2];
    const float* w2    = (const float*)d_in[3];
    const float* w3    = (const float*)d_in[4];
    const float* gamma = (const float*)d_in[5];
    const float* beta  = (const float*)d_in[6];
    const float* bias1 = (const float*)d_in[7];
    const float* alpha = (const float*)d_in[8];
    const float* bias2 = (const float*)d_in[9];

    char*  ws      = (char*)d_ws;
    float* scales  = (float*)(ws + WS_SCALES);
    float* gsum    = (float*)(ws + WS_GSUM);
    float* gsq     = (float*)(ws + WS_GSQ);
    float* bnscale = (float*)(ws + WS_BNSCALE);
    float* bnshift = (float*)(ws + WS_BNSHIFT);
    int*   prepB   = (int*)  (ws + WS_PREPB);
    char*  abuf    = (char*) (ws + WS_A);
    float* sbuf    = (float*)(ws + WS_S);

    prep_kernel <<<3,              256, 0, stream>>>(w1, w2, w3, scales, gsum, gsq, prepB);
    act_kernel  <<<BATCH * HH,     256, 0, stream>>>(x, bias0, abuf);
    gemm_kernel <<<(NPIX/128) * 4, 256, 0, stream>>>(abuf, prepB, scales, sbuf, gsum, gsq);
    bnfin_kernel<<<1,              256, 0, stream>>>(gsum, gsq, gamma, beta, bnscale, bnshift);
    out_kernel  <<<NPIX / 4,       256, 0, stream>>>(sbuf, x, bnscale, bnshift,
                                                     bias1, alpha, bias2, (float*)d_out);
}
